// Model_44813688766517
// MI455X (gfx1250) — compile-verified, matrix operand order checked by reading, not run on hardware
//
#include <hip/hip_runtime.h>
#include <hip/hip_bf16.h>
#include <math.h>

// ---------------------------------------------------------------------------
// Types for WMMA
// ---------------------------------------------------------------------------
typedef __attribute__((ext_vector_type(16))) __bf16 v16bf;
typedef __attribute__((ext_vector_type(8)))  float  v8f;

__device__ __forceinline__ unsigned short f2bf_rne(float f) {
    unsigned int u = __float_as_uint(f);
    unsigned int r = u + 0x7FFFu + ((u >> 16) & 1u);   // round-to-nearest-even
    return (unsigned short)(r >> 16);
}

// ---------------------------------------------------------------------------
// fp32 -> bf16 conversion (optionally fused ReLU). 8 elements per thread.
// ---------------------------------------------------------------------------
__global__ void convert_bf16(const float* __restrict__ in, unsigned short* __restrict__ out,
                             int n8, int relu) {
    const int id = blockIdx.x * blockDim.x + threadIdx.x;
    if (id >= n8) return;
    const float4* p = (const float4*)(in + (size_t)id * 8);
    float4 f0 = p[0], f1 = p[1];
    if (relu) {
        f0.x = fmaxf(f0.x, 0.f); f0.y = fmaxf(f0.y, 0.f);
        f0.z = fmaxf(f0.z, 0.f); f0.w = fmaxf(f0.w, 0.f);
        f1.x = fmaxf(f1.x, 0.f); f1.y = fmaxf(f1.y, 0.f);
        f1.z = fmaxf(f1.z, 0.f); f1.w = fmaxf(f1.w, 0.f);
    }
    union { unsigned short u[8]; uint4 q; } o;
    o.u[0] = f2bf_rne(f0.x); o.u[1] = f2bf_rne(f0.y);
    o.u[2] = f2bf_rne(f0.z); o.u[3] = f2bf_rne(f0.w);
    o.u[4] = f2bf_rne(f1.x); o.u[5] = f2bf_rne(f1.y);
    o.u[6] = f2bf_rne(f1.z); o.u[7] = f2bf_rne(f1.w);
    *(uint4*)(out + (size_t)id * 8) = o.q;
}

// ---------------------------------------------------------------------------
// Pack W[K,N] fp32 row-major into per-lane WMMA B-fragments (bf16):
//   Wp[(tileN*KC + kc)*512 + lane*16 + j], j = 2*v+p,
//   kb = (v>=4?16:0) + (lane>=16?8:0) + (v%4)*2
//   element = bf16( W[(kc*32 + kb + p)*N + tileN*16 + (lane&15)] )
// One thread per output ushort.
// ---------------------------------------------------------------------------
__global__ void pack_w_frag(const float* __restrict__ W, unsigned short* __restrict__ Wp,
                            int K, int N) {
    const int KC = K >> 5;
    const int total = (N >> 4) * KC * 512;
    const int tid = blockIdx.x * blockDim.x + threadIdx.x;
    if (tid >= total) return;
    const int j     = tid & 15;
    const int lane  = (tid >> 4) & 31;
    const int rest  = tid >> 9;
    const int kc    = rest % KC;
    const int tileN = rest / KC;
    const int v = j >> 1, p = j & 1;
    const int khalf = (lane >= 16) ? 8 : 0;
    const int kb = ((v & 4) << 2) + khalf + ((v & 3) << 1);
    const int row = kc * 32 + kb + p;
    const int col = tileN * 16 + (lane & 15);
    Wp[tid] = f2bf_rne(W[(size_t)row * N + col]);
}

// ---------------------------------------------------------------------------
// bf16 WMMA GEMM: C[M,N] = opt_relu( Abf[M,K] @ Wp + bias[N] )
// Abf: bf16 row-major (pre-converted). Wp: fragment-packed bf16 weights.
// One wave computes a 16x16 tile; inner loop = 4x b128 loads + 1 wmma.
// block = (32, GEMM_WAVES), grid = (N/16, ceil(M/16/GEMM_WAVES))
// ---------------------------------------------------------------------------
#define GEMM_WAVES 8

__global__ void gemm_bf16_wmma(const unsigned short* __restrict__ Abf,
                               const unsigned short* __restrict__ Wp,
                               const float* __restrict__ bias,
                               float* __restrict__ C,
                               int M, int K, int N, int relu_out) {
    const int lane  = threadIdx.x;                         // 0..31
    const int tileN = blockIdx.x;
    const int tileM = blockIdx.y * GEMM_WAVES + threadIdx.y;
    if (tileM * 16 >= M) return;

    const int mrow  = tileM * 16 + (lane & 15);
    const int ncol  = tileN * 16 + (lane & 15);
    const int khalf = (lane >> 4) << 3;                    // 0 or 8
    const int KC    = K >> 5;

    const unsigned short* arow  = Abf + (size_t)mrow * K + khalf;
    const unsigned short* wfrag = Wp + (size_t)tileN * KC * 512 + lane * 16;

    v8f acc = {};
    for (int kc = 0; kc < KC; ++kc) {
        union { v16bf v; uint4 q[2]; } a, b;
        a.q[0] = *(const uint4*)(arow + kc * 32);
        a.q[1] = *(const uint4*)(arow + kc * 32 + 16);
        b.q[0] = *(const uint4*)(wfrag + (size_t)kc * 512);
        b.q[1] = *(const uint4*)(wfrag + (size_t)kc * 512 + 8);
        acc = __builtin_amdgcn_wmma_f32_16x16x32_bf16(
            /*neg_a=*/false, a.v, /*neg_b=*/false, b.v,
            /*c_mod=*/(short)0, acc, /*reuse_a=*/false, /*reuse_b=*/false);
    }

    const float bs = bias ? bias[ncol] : 0.f;
#pragma unroll
    for (int r = 0; r < 8; ++r) {
        const int row = tileM * 16 + r + khalf;
        float val = acc[r] + bs;
        if (relu_out) val = fmaxf(val, 0.f);
        C[(size_t)row * N + ncol] = val;
    }
}

// ---------------------------------------------------------------------------
// Per-layer init: hout[n,k] = bias[k]; emax = -inf; denom = 0
// ---------------------------------------------------------------------------
__global__ void init_layer(float* __restrict__ emax, float* __restrict__ denom,
                           float* __restrict__ hout, const float* __restrict__ bias,
                           int Nn, int H) {
    const int id = blockIdx.x * blockDim.x + threadIdx.x;
    if (id < Nn * H) hout[id] = bias[id & (H - 1)];
    if (id < Nn) { emax[id] = -3.402823466e38f; denom[id] = 0.f; }
}

// ---------------------------------------------------------------------------
// Edge pass 1: e = att . leaky_relu(xl[src] + xr[dst]); atomic max into emax[dst]
// ---------------------------------------------------------------------------
__device__ __forceinline__ void atomic_max_f32(float* addr, float val) {
#if __has_builtin(__hip_atomic_fetch_max)
    __hip_atomic_fetch_max(addr, val, __ATOMIC_RELAXED, __HIP_MEMORY_SCOPE_AGENT);
#else
    unsigned int* ua = (unsigned int*)addr;
    unsigned int old = *ua;
    while (__uint_as_float(old) < val) {
        unsigned int assumed = old;
        old = atomicCAS(ua, assumed, __float_as_uint(val));
        if (old == assumed) break;
    }
#endif
}

__global__ void edge_score(const float* __restrict__ xl, const float* __restrict__ xr,
                           const float* __restrict__ att,
                           const int* __restrict__ src, const int* __restrict__ dst,
                           int E, int Nn, float* __restrict__ e, float* __restrict__ emax) {
    const int id = blockIdx.x * blockDim.x + threadIdx.x;
    if (id >= E + Nn) return;
    int s, d;
    if (id < E) { s = src[id]; d = dst[id]; } else { s = d = id - E; }

    const float4* pl = (const float4*)(xl + (size_t)s * 128);
    const float4* pr = (const float4*)(xr + (size_t)d * 128);
    const float4* pa = (const float4*)att;
    float acc = 0.f;
#pragma unroll 8
    for (int i = 0; i < 32; ++i) {
        const float4 a = pl[i], b = pr[i], w = pa[i];
        float m;
        m = a.x + b.x; m = (m > 0.f) ? m : 0.2f * m; acc += m * w.x;
        m = a.y + b.y; m = (m > 0.f) ? m : 0.2f * m; acc += m * w.y;
        m = a.z + b.z; m = (m > 0.f) ? m : 0.2f * m; acc += m * w.z;
        m = a.w + b.w; m = (m > 0.f) ? m : 0.2f * m; acc += m * w.w;
    }
    e[id] = acc;
    atomic_max_f32(&emax[d], acc);
}

// ---------------------------------------------------------------------------
// Edge pass 2: ee = exp(e - emax[dst]); denom[dst] += ee  (in-place on e)
// ---------------------------------------------------------------------------
__global__ void edge_exp(const int* __restrict__ dst, int E, int Nn,
                         float* __restrict__ e, const float* __restrict__ emax,
                         float* __restrict__ denom) {
    const int id = blockIdx.x * blockDim.x + threadIdx.x;
    if (id >= E + Nn) return;
    const int d = (id < E) ? dst[id] : (id - E);
    const float ee = __expf(e[id] - emax[d]);
    e[id] = ee;
    atomicAdd(&denom[d], ee);
}

// ---------------------------------------------------------------------------
// Edge pass 3: out[dst] += (ee/denom[dst]) * xl[src]
// One wave per edge; f32 atomics resolve in L2 (all buffers L2-resident).
// ---------------------------------------------------------------------------
__global__ void edge_scatter(const float* __restrict__ xl, const float* __restrict__ e,
                             const float* __restrict__ denom,
                             const int* __restrict__ src, const int* __restrict__ dst,
                             int E, int Nn, float* __restrict__ out) {
    const size_t tid  = (size_t)blockIdx.x * blockDim.x + threadIdx.x;
    const size_t edge = tid >> 5;
    const int    part = (int)(tid & 31);
    if (edge >= (size_t)(E + Nn)) return;
    int s, d;
    if (edge < (size_t)E) { s = src[edge]; d = dst[edge]; } else { s = d = (int)edge - E; }

    const float alpha = e[edge] / denom[d];
    const float4 v = ((const float4*)(xl + (size_t)s * 128))[part];
    float* o = out + (size_t)d * 128 + part * 4;
    atomicAdd(o + 0, alpha * v.x);
    atomicAdd(o + 1, alpha * v.y);
    atomicAdd(o + 2, alpha * v.z);
    atomicAdd(o + 3, alpha * v.w);
}

// ---------------------------------------------------------------------------
// Pooling
// ---------------------------------------------------------------------------
__global__ void zero_f32(float* __restrict__ p, int n) {
    const int id = blockIdx.x * blockDim.x + threadIdx.x;
    if (id < n) p[id] = 0.f;
}

__global__ void pool_sum(const float* __restrict__ h, const int* __restrict__ batch,
                         float* __restrict__ sums, float* __restrict__ cnts, int Nn) {
    const size_t tid  = (size_t)blockIdx.x * blockDim.x + threadIdx.x;
    const size_t node = tid >> 5;
    const int    part = (int)(tid & 31);
    if (node >= (size_t)Nn) return;
    const int b = batch[node];
    const float4 v = ((const float4*)(h + node * 128))[part];
    float* s = sums + (size_t)b * 128 + part * 4;
    atomicAdd(s + 0, v.x);
    atomicAdd(s + 1, v.y);
    atomicAdd(s + 2, v.z);
    atomicAdd(s + 3, v.w);
    if (part == 0) atomicAdd(&cnts[b], 1.0f);
}

// ---------------------------------------------------------------------------
// feats[b] = [ mean-pool | day_emb | hour_emb ]
// ---------------------------------------------------------------------------
__global__ void build_feats(const float* __restrict__ sums, const float* __restrict__ cnts,
                            const int* __restrict__ day, const int* __restrict__ hour,
                            const float* __restrict__ day_tab, const float* __restrict__ hour_tab,
                            float* __restrict__ feats) {
    const int b = blockIdx.x;
    const int t = threadIdx.x;   // 0..255
    float v;
    if (t < 128)      v = sums[b * 128 + t] / fmaxf(cnts[b], 1.0f);
    else if (t < 192) v = day_tab[day[b] * 64 + (t - 128)];
    else              v = hour_tab[hour[b] * 64 + (t - 192)];
    feats[b * 256 + t] = v;
}

// ---------------------------------------------------------------------------
// Final projection: out[b] = z2[b,:] . W3 + bl3
// ---------------------------------------------------------------------------
__global__ void final_dot(const float* __restrict__ z2, const float* __restrict__ W3,
                          const float* __restrict__ bl3, float* __restrict__ out, int B) {
    const int b = blockIdx.x * blockDim.x + threadIdx.x;
    if (b >= B) return;
    float acc = bl3[0];
#pragma unroll 8
    for (int k = 0; k < 256; ++k) acc += z2[b * 256 + k] * W3[k];
    out[b] = acc;
}

// ---------------------------------------------------------------------------
// Host-side orchestration
// ---------------------------------------------------------------------------
static inline size_t align256(size_t x) { return (x + 255) & ~(size_t)255; }

static void launch_gemm(const unsigned short* Abf, const unsigned short* Wp,
                        const float* bias, float* C, int M, int K, int N,
                        int relu_out, hipStream_t stream) {
    dim3 block(32, GEMM_WAVES);
    dim3 grid(N / 16, (M / 16 + GEMM_WAVES - 1) / GEMM_WAVES);
    gemm_bf16_wmma<<<grid, block, 0, stream>>>(Abf, Wp, bias, C, M, K, N, relu_out);
}

static void run_gat_layer(const float* xin, int relu_in,
                          const float* Wl, const float* Wr,
                          const float* att, const float* bias,
                          unsigned short* Abf, unsigned short* WpA, unsigned short* WpB,
                          float* xl, float* xr, float* e, float* emax, float* denom,
                          float* hout,
                          const int* src, const int* dst,
                          int Nn, int E, hipStream_t stream) {
    const int H = 128, K = 128;
    // activations -> bf16 (ReLU fused), weights -> fragment-packed bf16
    {
        int n8 = Nn * K / 8;
        convert_bf16<<<(n8 + 255) / 256, 256, 0, stream>>>(xin, Abf, n8, relu_in);
        int wtot = (H / 16) * (K / 32) * 512;
        pack_w_frag<<<(wtot + 255) / 256, 256, 0, stream>>>(Wl, WpA, K, H);
        pack_w_frag<<<(wtot + 255) / 256, 256, 0, stream>>>(Wr, WpB, K, H);
    }
    launch_gemm(Abf, WpA, nullptr, xl, Nn, K, H, 0, stream);
    launch_gemm(Abf, WpB, nullptr, xr, Nn, K, H, 0, stream);

    // init emax/denom/hout(bias)
    {
        int n = Nn * H;
        init_layer<<<(n + 255) / 256, 256, 0, stream>>>(emax, denom, hout, bias, Nn, H);
    }
    const int ET = E + Nn;  // with self loops
    edge_score<<<(ET + 255) / 256, 256, 0, stream>>>(xl, xr, att, src, dst, E, Nn, e, emax);
    edge_exp<<<(ET + 255) / 256, 256, 0, stream>>>(dst, E, Nn, e, emax, denom);
    {
        size_t nthreads = (size_t)ET * 32;
        edge_scatter<<<(unsigned)((nthreads + 255) / 256), 256, 0, stream>>>(
            xl, e, denom, src, dst, E, Nn, hout);
    }
}

extern "C" void kernel_launch(void* const* d_in, const int* in_sizes, int n_in,
                              void* d_out, int out_size, void* d_ws, size_t ws_size,
                              hipStream_t stream) {
    // --- inputs (setup_inputs order) ---
    const float* x       = (const float*)d_in[0];
    const int*   ei      = (const int*)  d_in[1];
    const int*   batch   = (const int*)  d_in[2];
    const int*   day     = (const int*)  d_in[3];
    const int*   hour    = (const int*)  d_in[4];
    const float* Wl1 = (const float*)d_in[5];  const float* Wr1 = (const float*)d_in[6];
    const float* att1= (const float*)d_in[7];  const float* b1  = (const float*)d_in[8];
    const float* Wl2 = (const float*)d_in[9];  const float* Wr2 = (const float*)d_in[10];
    const float* att2= (const float*)d_in[11]; const float* b2  = (const float*)d_in[12];
    const float* Wl3 = (const float*)d_in[13]; const float* Wr3 = (const float*)d_in[14];
    const float* att3= (const float*)d_in[15]; const float* b3  = (const float*)d_in[16];
    const float* day_tab  = (const float*)d_in[17];
    const float* hour_tab = (const float*)d_in[18];
    const float* W1  = (const float*)d_in[19]; const float* bl1 = (const float*)d_in[20];
    const float* W2  = (const float*)d_in[21]; const float* bl2 = (const float*)d_in[22];
    const float* W3  = (const float*)d_in[23]; const float* bl3 = (const float*)d_in[24];

    const int Nn = in_sizes[2];          // 40000 nodes
    const int E  = in_sizes[1] / 2;      // 640000 edges
    const int B  = in_sizes[3];          // 64 graphs
    const int*   src = ei;
    const int*   dst = ei + E;
    float* out = (float*)d_out;

    // --- workspace carve-up ---
    char* p = (char*)d_ws;
    size_t off = 0;
    const size_t szNH = (size_t)Nn * 128 * sizeof(float);
    float* xl    = (float*)(p + off); off = align256(off + szNH);
    float* xr    = (float*)(p + off); off = align256(off + szNH);
    float* hA    = (float*)(p + off); off = align256(off + szNH);
    float* hB    = (float*)(p + off); off = align256(off + szNH);
    float* e     = (float*)(p + off); off = align256(off + (size_t)(E + Nn) * sizeof(float));
    float* emax  = (float*)(p + off); off = align256(off + (size_t)Nn * sizeof(float));
    float* denom = (float*)(p + off); off = align256(off + (size_t)Nn * sizeof(float));
    float* sums  = (float*)(p + off); off = align256(off + (size_t)B * 128 * sizeof(float));
    float* cnts  = (float*)(p + off); off = align256(off + (size_t)B * sizeof(float));
    float* feats = (float*)(p + off); off = align256(off + (size_t)B * 256 * sizeof(float));
    float* z1    = (float*)(p + off); off = align256(off + (size_t)B * 256 * sizeof(float));
    float* z2    = (float*)(p + off); off = align256(off + (size_t)B * 256 * sizeof(float));
    // bf16 staging: activations (max of Nn*128 and B*256 elements) + packed weights
    unsigned short* Abf = (unsigned short*)(p + off);
    off = align256(off + (size_t)Nn * 128 * sizeof(unsigned short));
    unsigned short* WpA = (unsigned short*)(p + off);
    off = align256(off + (size_t)256 * 256 * sizeof(unsigned short));
    unsigned short* WpB = (unsigned short*)(p + off);
    off = align256(off + (size_t)256 * 256 * sizeof(unsigned short));
    (void)ws_size;

    // --- GAT layers (relu fused into next layer's bf16 conversion) ---
    run_gat_layer(x,  0, Wl1, Wr1, att1, b1, Abf, WpA, WpB,
                  xl, xr, e, emax, denom, hA, src, dst, Nn, E, stream);
    run_gat_layer(hA, 1, Wl2, Wr2, att2, b2, Abf, WpA, WpB,
                  xl, xr, e, emax, denom, hB, src, dst, Nn, E, stream);
    run_gat_layer(hB, 1, Wl3, Wr3, att3, b3, Abf, WpA, WpB,
                  xl, xr, e, emax, denom, hA, src, dst, Nn, E, stream);

    // --- global mean pool over graphs ---
    {
        zero_f32<<<(B * 128 + 255) / 256, 256, 0, stream>>>(sums, B * 128);
        zero_f32<<<1, 256, 0, stream>>>(cnts, B);
        size_t nthreads = (size_t)Nn * 32;
        pool_sum<<<(unsigned)((nthreads + 255) / 256), 256, 0, stream>>>(hA, batch, sums, cnts, Nn);
    }

    // --- feats + MLP head (WMMA GEMMs on pre-packed bf16) ---
    build_feats<<<B, 256, 0, stream>>>(sums, cnts, day, hour, day_tab, hour_tab, feats);
    {
        int n8 = B * 256 / 8;
        int wtot = (256 / 16) * (256 / 32) * 512;
        convert_bf16<<<(n8 + 255) / 256, 256, 0, stream>>>(feats, Abf, n8, 0);
        pack_w_frag<<<(wtot + 255) / 256, 256, 0, stream>>>(W1, WpA, 256, 256);
        launch_gemm(Abf, WpA, bl1, z1, B, 256, 256, 1, stream);

        convert_bf16<<<(n8 + 255) / 256, 256, 0, stream>>>(z1, Abf, n8, 0);
        pack_w_frag<<<(wtot + 255) / 256, 256, 0, stream>>>(W2, WpB, 256, 256);
        launch_gemm(Abf, WpB, bl2, z2, B, 256, 256, 1, stream);
    }
    final_dot<<<1, 64, 0, stream>>>(z2, W3, bl3, out, B);
}
